// SelfAttention_84808424227161
// MI455X (gfx1250) — matrix operand order repaired and checked by reading, not compiled, not run on hardware
//
#include <hip/hip_runtime.h>
#include <hip/hip_bf16.h>

// MI455X / gfx1250: bf16 WMMA (v_wmma_f32_16x16x32_bf16) for all matmuls,
// double-buffered LDS tiles, async global->LDS copies (ASYNCcnt-tracked),
// flash-attention with transposed-score trick (wave32 layouts per CDNA5 ISA
// 7.12.2).

typedef __bf16 bf16_t;
typedef bf16_t bf16x4  __attribute__((ext_vector_type(4)));
typedef bf16_t bf16x8  __attribute__((ext_vector_type(8)));
typedef bf16_t bf16x16 __attribute__((ext_vector_type(16)));
typedef float  f32x8   __attribute__((ext_vector_type(8)));
typedef int    v4i     __attribute__((ext_vector_type(4)));

#define DEV __device__ __forceinline__

#if defined(__has_builtin)
#  if __has_builtin(__builtin_amdgcn_global_load_async_to_lds_b128) && \
      __has_builtin(__builtin_amdgcn_s_wait_asynccnt)
#    define HAVE_ASYNC_LDS 1
#  endif
#endif
#ifndef HAVE_ASYNC_LDS
#  define HAVE_ASYNC_LDS 0
#endif

#if HAVE_ASYNC_LDS
typedef __attribute__((address_space(1))) v4i as1_v4i;  // global (HIP: __device__)
typedef __attribute__((address_space(3))) v4i as3_v4i;  // LDS    (HIP: __shared__)
// 16B global -> LDS, tracked on ASYNCcnt (CDNA5 §10 async ops).
DEV void async_cp16(const bf16_t* g, bf16_t* l) {
  __builtin_amdgcn_global_load_async_to_lds_b128((as1_v4i*)g, (as3_v4i*)l, 0, 0);
}
DEV void async_wait0() { __builtin_amdgcn_s_wait_asynccnt(0); }
#else
DEV void async_wait0() {}
#endif

// A-fragment (16x32, 16-bit): lane l holds row M=l&15; element e holds
// K = e + 16*(l>=16)  -> two 16B loads at +16h and +16h+8.
DEV bf16x16 frag_a(const bf16_t* prow, int hh) {
  const bf16_t* p = prow + 16 * hh;
  bf16x8 lo = *(const bf16x8*)p;
  bf16x8 hi = *(const bf16x8*)(p + 8);
  return __builtin_shufflevector(lo, hi, 0,1,2,3,4,5,6,7,8,9,10,11,12,13,14,15);
}

// B-fragment (32x16, 16-bit): lane l holds column N=l&15; element e holds
// K = e + 16*(l>=16)  -> 32 contiguous bytes at +16h.
DEV bf16x16 frag_b(const bf16_t* prow, int hh) {
  const bf16_t* p = prow + 16 * hh;
  bf16x8 lo = *(const bf16x8*)p;
  bf16x8 hi = *(const bf16x8*)(p + 8);
  return __builtin_shufflevector(lo, hi, 0,1,2,3,4,5,6,7,8,9,10,11,12,13,14,15);
}

DEV f32x8 wmma_bf16(bf16x16 a, bf16x16 b, f32x8 c) {
  // 8 args: (neg_a, A, neg_b, B, c_mod, C, reuse_a, reuse_b)
  return __builtin_amdgcn_wmma_f32_16x16x32_bf16(false, a, false, b, (short)0, c,
                                                 false, false);
}

// ---------------------------------------------------------------- fp32->bf16
__global__ __launch_bounds__(256)
void cvt_f32_bf16(const float* __restrict__ in, bf16_t* __restrict__ out, int n4) {
  int i = blockIdx.x * 256 + threadIdx.x;
  if (i >= n4) return;
  float4 f = ((const float4*)in)[i];
  bf16x4 o;
  o.x = (bf16_t)f.x; o.y = (bf16_t)f.y; o.z = (bf16_t)f.z; o.w = (bf16_t)f.w;
  ((bf16x4*)out)[i] = o;
}

// ------------------------------------------------------------------- GEMM TN
// C[M][N] = A[M][K] * W[N][K]^T, A/W bf16 row-major over K, C = OutT.
// Block: 128x128 tile, 256 threads = 8 waves (4 along M x 2 along N), each
// wave computes 32x64 as 2x4 WMMA accumulators. K-step 64, double-buffered
// LDS (one barrier per 16 WMMAs, next tile loads overlap current compute).
template <typename OutT>
__global__ __launch_bounds__(256)
void gemm_tn(const bf16_t* __restrict__ A, const bf16_t* __restrict__ W,
             OutT* __restrict__ C, int M, int N, int K) {
  __shared__ bf16_t As[2][128][72];  // stride 72 bf16 = 144B: 16B-aligned
  __shared__ bf16_t Bs[2][128][72];

  const int tid = threadIdx.x;
  const int lane = tid & 31;
  const int w = tid >> 5;
  const int wm = w & 3, wn = w >> 2;
  const int m0 = blockIdx.y * 128, n0 = blockIdx.x * 128;
  const int cl = lane & 15, hh = lane >> 4;
  const int lrow = tid >> 1, lcol = (tid & 1) * 32;

  f32x8 acc[2][4] = {};

  const bf16_t* ag = A + (size_t)(m0 + lrow) * K + lcol;
  const bf16_t* bg = W + (size_t)(n0 + lrow) * K + lcol;

  auto load_tile = [&](int buf, int k0) {
#if HAVE_ASYNC_LDS
#pragma unroll
    for (int s = 0; s < 4; ++s) {
      async_cp16(ag + k0 + s * 8, &As[buf][lrow][lcol + s * 8]);
      async_cp16(bg + k0 + s * 8, &Bs[buf][lrow][lcol + s * 8]);
    }
#else
#pragma unroll
    for (int s = 0; s < 4; ++s) {
      *(bf16x8*)&As[buf][lrow][lcol + s * 8] = *(const bf16x8*)(ag + k0 + s * 8);
      *(bf16x8*)&Bs[buf][lrow][lcol + s * 8] = *(const bf16x8*)(bg + k0 + s * 8);
    }
    if (k0 + 64 < K) {
      __builtin_prefetch(ag + k0 + 64, 0, 1);   // global_prefetch_b8
      __builtin_prefetch(bg + k0 + 64, 0, 1);
    }
#endif
  };

  load_tile(0, 0);
  async_wait0();
  __syncthreads();

  const int ntiles = K >> 6;
  for (int t = 0; t < ntiles; ++t) {
    const int cur = t & 1;
    if (t + 1 < ntiles) load_tile(1 - cur, (t + 1) << 6);  // into other buffer

#pragma unroll
    for (int ks = 0; ks < 2; ++ks) {
      bf16x16 af[2], bfr[4];
#pragma unroll
      for (int mt = 0; mt < 2; ++mt)
        af[mt] = frag_a(&As[cur][wm * 32 + mt * 16 + cl][ks * 32], hh);
#pragma unroll
      for (int nt = 0; nt < 4; ++nt)
        bfr[nt] = frag_b(&Bs[cur][wn * 64 + nt * 16 + cl][ks * 32], hh);
#pragma unroll
      for (int mt = 0; mt < 2; ++mt)
#pragma unroll
        for (int nt = 0; nt < 4; ++nt)
          acc[mt][nt] = wmma_bf16(af[mt], bfr[nt], acc[mt][nt]);
    }
    async_wait0();      // next-tile async copies done before barrier
    __syncthreads();
  }

  // C/D layout: lane = n + 16*(m>=8), VGPR v -> m = v + 8*(lane>=16)
#pragma unroll
  for (int mt = 0; mt < 2; ++mt)
#pragma unroll
    for (int nt = 0; nt < 4; ++nt) {
      int col = n0 + wn * 64 + nt * 16 + cl;
#pragma unroll
      for (int v = 0; v < 8; ++v) {
        int row = m0 + wm * 32 + mt * 16 + v + 8 * hh;
        C[(size_t)row * N + col] = (OutT)acc[mt][nt][v];
      }
    }
}

// ---------------------------------------------------------------------- RoPE
__global__ __launch_bounds__(256)
void rope_bf16(bf16_t* __restrict__ xb, int nheads, int rowdim,
               const int* __restrict__ start_pos) {
  int i = blockIdx.x * 256 + threadIdx.x;  // grid sized exactly
  int pair = i & 63;
  int rest = i >> 6;
  int h = rest % nheads;
  int tok = rest / nheads;        // 0..4095 (b*2048 + s)
  int s = tok & 2047;
  float pos = (float)(start_pos[0] + s);
  float inv_freq = __powf(10000.0f, -(float)(2 * pair) * (1.0f / 128.0f));
  float ang = pos * inv_freq;
  float sn, cs;
  __sincosf(ang, &sn, &cs);
  bf16_t* p = xb + (size_t)tok * rowdim + h * 128 + pair * 2;
  float x1 = (float)p[0], x2 = (float)p[1];
  p[0] = (bf16_t)(x1 * cs - x2 * sn);
  p[1] = (bf16_t)(x1 * sn + x2 * cs);
}

// ----------------------------------------------------------- flash attention
// Block = (32 q-rows, one kv-head, one batch); 8 waves = 4 rep-heads x 2
// q-subtiles of 16 rows; K/V tiles (32 keys) double-buffered in LDS, shared
// across the 4 GQA heads. Scores computed transposed: S^T = K_tile(A) x
// Q^T(B) so softmax stats are per-lane scalars (q = lane&15); then
// O^T = V^T(A) x P^T(B) with P built from score accumulators via one
// shfl_xor(16) half exchange.
__global__ __launch_bounds__(256)
void attn_kernel(const bf16_t* __restrict__ qb, const bf16_t* __restrict__ kb,
                 const bf16_t* __restrict__ vb, bf16_t* __restrict__ ctx) {
  __shared__ bf16_t Ks[2][32][136];   // [key][dim], stride 136 bf16
  __shared__ bf16_t VTs[2][128][40];  // [dim][key], transposed V tile

  const int tid = threadIdx.x, lane = tid & 31, w = tid >> 5;
  const int qsub = w & 1, rep = w >> 1;
  const int xblk = blockIdx.x, kvh = blockIdx.y, b = blockIdx.z;
  const int tokBase = b * 2048 + xblk * 32 + qsub * 16;
  const int qhead = kvh * 4 + rep;
  const int cl = lane & 15, hh = lane >> 4;

  // Preload Q^T B-fragments (dim chunks of 32), straight from global.
  bf16x16 qf[4];
  {
    const bf16_t* qrow = qb + (size_t)(tokBase + cl) * 4096 + qhead * 128;
#pragma unroll
    for (int ch = 0; ch < 4; ++ch) qf[ch] = frag_b(qrow + ch * 32, hh);
  }

  f32x8 o[8] = {};                 // O^T accumulators, 8 dim-tiles of 16
  float mrow = -1e30f, lsum = 0.0f;
  const float scale = 0.08838834764831845f;  // 1/sqrt(128)

  const int lkey = tid >> 3;            // 0..31
  const int lseg = (tid & 7) * 8;       // dim segment (8 bf16)
  const bf16_t* kg = kb + (size_t)(b * 2048 + lkey) * 1024 + kvh * 128 + lseg;
  const bf16_t* vg = vb + (size_t)(b * 2048 + lkey) * 1024 + kvh * 128 + lseg;

  auto load_kv = [&](int buf, int key0) {
    size_t gofs = (size_t)key0 * 1024;
#if HAVE_ASYNC_LDS
    async_cp16(kg + gofs,      &Ks[buf][lkey][lseg]);
    async_cp16(kg + gofs + 64, &Ks[buf][lkey][lseg + 64]);
#else
    *(bf16x8*)&Ks[buf][lkey][lseg]      = *(const bf16x8*)(kg + gofs);
    *(bf16x8*)&Ks[buf][lkey][lseg + 64] = *(const bf16x8*)(kg + gofs + 64);
#endif
    bf16x8 v0 = *(const bf16x8*)(vg + gofs);       // V needs a transpose:
    bf16x8 v1 = *(const bf16x8*)(vg + gofs + 64);  // stage through VGPRs
#pragma unroll
    for (int j = 0; j < 8; ++j) {
      VTs[buf][lseg + j][lkey]      = v0[j];
      VTs[buf][lseg + 64 + j][lkey] = v1[j];
    }
  };

  load_kv(0, 0);
  async_wait0();
  __syncthreads();

  for (int it = 0; it < 64; ++it) {
    const int cur = it & 1;
    if (it + 1 < 64) load_kv(1 - cur, (it + 1) * 32);  // other buffer

    // S^T tiles: c0 = keys 0-15, c1 = keys 16-31 (each 16keys x 16q).
    // Fragment loads grouped so ds_loads clause up ahead of the WMMAs.
    f32x8 c0 = {}, c1 = {};
#pragma unroll
    for (int chp = 0; chp < 4; chp += 2) {
      bf16x16 kf0 = frag_a(&Ks[cur][cl][chp * 32], hh);
      bf16x16 kf1 = frag_a(&Ks[cur][16 + cl][chp * 32], hh);
      bf16x16 kf2 = frag_a(&Ks[cur][cl][(chp + 1) * 32], hh);
      bf16x16 kf3 = frag_a(&Ks[cur][16 + cl][(chp + 1) * 32], hh);
      c0 = wmma_bf16(kf0, qf[chp], c0);
      c1 = wmma_bf16(kf1, qf[chp], c1);
      c0 = wmma_bf16(kf2, qf[chp + 1], c0);
      c1 = wmma_bf16(kf3, qf[chp + 1], c1);
    }

    // Online softmax; this lane owns q = cl. Its 16 values cover keys
    // {v+8h} (c0) and {16+v+8h} (c1); the other half-lane has the rest.
    float mx = -1e30f;
#pragma unroll
    for (int v = 0; v < 8; ++v) { mx = fmaxf(mx, c0[v]); mx = fmaxf(mx, c1[v]); }
    mx *= scale;
    mx = fmaxf(mx, __shfl_xor(mx, 16));
    float mnew = fmaxf(mrow, mx);
    float alpha = __expf(mrow - mnew);

    float p0[8], p1[8], ps = 0.0f;
#pragma unroll
    for (int v = 0; v < 8; ++v) {
      p0[v] = __expf(c0[v] * scale - mnew); ps += p0[v];
      p1[v] = __expf(c1[v] * scale - mnew); ps += p1[v];
    }
    ps += __shfl_xor(ps, 16);
    lsum = lsum * alpha + ps;
    mrow = mnew;
#pragma unroll
    for (int t = 0; t < 8; ++t)
#pragma unroll
      for (int v = 0; v < 8; ++v) o[t][v] *= alpha;

    // Build P^T B-fragment: element e needs key = e + 16*hh.
    // h=0 needs keys 0-15 (own p0 + other-half p0); h=1 keys 16-31.
    float p0x[8], p1x[8];
#pragma unroll
    for (int v = 0; v < 8; ++v) {
      p0x[v] = __shfl_xor(p0[v], 16);
      p1x[v] = __shfl_xor(p1[v], 16);
    }
    bf16x16 pf;
#pragma unroll
    for (int e = 0; e < 8; ++e) {
      pf[e]     = (bf16_t)(hh ? p1x[e] : p0[e]);
      pf[e + 8] = (bf16_t)(hh ? p1[e]  : p0x[e]);
    }

    // O^T += V^T x P^T, fragment loads grouped 4-at-a-time.
#pragma unroll
    for (int dtp = 0; dtp < 8; dtp += 4) {
      bf16x16 vf0 = frag_a(&VTs[cur][(dtp + 0) * 16 + cl][0], hh);
      bf16x16 vf1 = frag_a(&VTs[cur][(dtp + 1) * 16 + cl][0], hh);
      bf16x16 vf2 = frag_a(&VTs[cur][(dtp + 2) * 16 + cl][0], hh);
      bf16x16 vf3 = frag_a(&VTs[cur][(dtp + 3) * 16 + cl][0], hh);
      o[dtp + 0] = wmma_bf16(vf0, pf, o[dtp + 0]);
      o[dtp + 1] = wmma_bf16(vf1, pf, o[dtp + 1]);
      o[dtp + 2] = wmma_bf16(vf2, pf, o[dtp + 2]);
      o[dtp + 3] = wmma_bf16(vf3, pf, o[dtp + 3]);
    }
    async_wait0();
    __syncthreads();
  }

  float inv = 1.0f / lsum;
  bf16_t* crow = ctx + (size_t)(tokBase + cl) * 4096 + qhead * 128 + 8 * hh;
#pragma unroll
  for (int dt = 0; dt < 8; ++dt) {
    bf16x8 pk;
#pragma unroll
    for (int v = 0; v < 8; ++v) pk[v] = (bf16_t)(o[dt][v] * inv);
    *(bf16x8*)(crow + dt * 16) = pk;   // dims dt*16 + 8*hh + v, contiguous 16B
  }
}

// -------------------------------------------------------------------- launch
extern "C" void kernel_launch(void* const* d_in, const int* in_sizes, int n_in,
                              void* d_out, int out_size, void* d_ws, size_t ws_size,
                              hipStream_t stream) {
  (void)in_sizes; (void)n_in; (void)out_size; (void)ws_size;
  const float* x  = (const float*)d_in[0];
  const float* wq = (const float*)d_in[1];
  const float* wk = (const float*)d_in[2];
  const float* wv = (const float*)d_in[3];
  const float* wo = (const float*)d_in[4];
  const int* start_pos = (const int*)d_in[5];
  float* out = (float*)d_out;

  char* p = (char*)d_ws;
  auto take = [&](size_t nelem) {
    bf16_t* r = (bf16_t*)p;
    p += nelem * sizeof(bf16_t);
    return r;
  };
  bf16_t* xb   = take(16777216);  // [4096][4096]
  bf16_t* wqb  = take(16777216);
  bf16_t* wkb  = take(4194304);   // [1024][4096]
  bf16_t* wvb  = take(4194304);
  bf16_t* wob  = take(16777216);
  bf16_t* qbuf = take(16777216);  // [4096][4096]
  bf16_t* kbuf = take(4194304);   // [4096][1024]
  bf16_t* vbuf = take(4194304);
  bf16_t* ctx  = take(16777216);

  cvt_f32_bf16<<<16384, 256, 0, stream>>>(x,  xb,  4194304);
  cvt_f32_bf16<<<16384, 256, 0, stream>>>(wq, wqb, 4194304);
  cvt_f32_bf16<<<4096,  256, 0, stream>>>(wk, wkb, 1048576);
  cvt_f32_bf16<<<4096,  256, 0, stream>>>(wv, wvb, 1048576);
  cvt_f32_bf16<<<16384, 256, 0, stream>>>(wo, wob, 4194304);

  gemm_tn<bf16_t><<<dim3(32, 32), 256, 0, stream>>>(xb, wqb, qbuf, 4096, 4096, 4096);
  gemm_tn<bf16_t><<<dim3(8, 32),  256, 0, stream>>>(xb, wkb, kbuf, 4096, 1024, 4096);
  gemm_tn<bf16_t><<<dim3(8, 32),  256, 0, stream>>>(xb, wvb, vbuf, 4096, 1024, 4096);

  rope_bf16<<<32768, 256, 0, stream>>>(qbuf, 32, 4096, start_pos);
  rope_bf16<<<8192,  256, 0, stream>>>(kbuf, 8, 1024, start_pos);

  attn_kernel<<<dim3(64, 8, 2), 256, 0, stream>>>(qbuf, kbuf, vbuf, ctx);

  gemm_tn<float><<<dim3(32, 32), 256, 0, stream>>>(ctx, wob, out, 4096, 4096, 4096);
}
